// TransformerEncoderLayer_65274912965001
// MI455X (gfx1250) — matrix operand order repaired and checked by reading, not compiled, not run on hardware
//
#include <hip/hip_runtime.h>
#include <cstdint>
#include <cstddef>

typedef _Float16 f16;
typedef _Float16 v16h __attribute__((ext_vector_type(16)));
typedef _Float16 v8h  __attribute__((ext_vector_type(8)));
typedef float    v8f  __attribute__((ext_vector_type(8)));
typedef float    v4f  __attribute__((ext_vector_type(4)));

#define NB    4
#define TDIM  2048
#define BDIM  768
#define NH    12
#define HDIM  64
#define DFF   3072
#define MROWS (NB*TDIM)   /* 8192 */

// Epilogue / variant flags
enum { F_AF32 = 1, F_BIAS = 2, F_RESID = 4, F_GELU = 8, F_SCALE = 16, F_STF32 = 32, F_STF16 = 64 };

#define LDS_TILE (128 * 40)          /* f16 elements per buffer */
#define LDS_TILE_BYTES (LDS_TILE * 2)

// Async global->LDS copy of 2x16B per lane (CDNA5 GLOBAL_LOAD_ASYNC_TO_LDS_B128).
// INST_OFFSET is added to both the LDS and global address, so the second chunk
// reuses the same operands with offset:16.
__device__ __forceinline__ void async_copy_32B(unsigned ldsOff, const void* gsrc)
{
    asm volatile("global_load_async_to_lds_b128 %0, %1, off\n\t"
                 "global_load_async_to_lds_b128 %0, %1, off offset:16"
                 :: "v"(ldsOff), "v"(gsrc) : "memory");
}

// ---------------------------------------------------------------------------
// Generic batched WMMA GEMM:  C[M,N] = A[M,K] * Bt[N,K]^T  (Bt stored N-major)
// 256 threads = 8 waves (wave32). BM=128, BN=128, BK=32, double-buffered LDS,
// async global->LDS staging (ASYNCcnt), v_wmma_f32_16x16x32_f16 compute.
// Wave grid 4(M) x 2(N); each wave owns 2x4 tiles of 16x16 -> 32 rows x 64 cols.
// Batch (blockIdx.z) decomposed as b = z/Hh, h = z%Hh with independent strides.
// ---------------------------------------------------------------------------
template<int FLAGS>
__global__ __launch_bounds__(256)
void gemm_wmma(const void* __restrict__ Av, long long sAb, long long sAh, int lda,
               const f16* __restrict__ Bt, long long sBb, long long sBh, int ldb,
               const float* __restrict__ bias,
               const float* __restrict__ resid, long long sRb, long long sRh, int ldr,
               float* __restrict__ outF, f16* __restrict__ outH,
               long long sCb, long long sCh, int ldc,
               int N, int Kd, int Hh, float scale)
{
    __shared__ f16 As[2][LDS_TILE];   // [row][k] padded stride 40
    __shared__ f16 Bs[2][LDS_TILE];   // [col][k] padded stride 40 (N-major)

    const int tid  = threadIdx.x;
    const int lane = tid & 31;
    const int wid  = tid >> 5;
    const int wM   = wid >> 1;         // 0..3
    const int wN   = wid & 1;          // 0..1

    const int bh = blockIdx.z;
    const int bb = bh / Hh, hh = bh % Hh;
    const long long aOff = (long long)bb * sAb + (long long)hh * sAh;
    const long long bOff = (long long)bb * sBb + (long long)hh * sBh;
    const long long cOff = (long long)bb * sCb + (long long)hh * sCh;
    const long long rOff = (long long)bb * sRb + (long long)hh * sRh;

    const int tileM = blockIdx.y * 128;
    const int tileN = blockIdx.x * 128;

    const int srow  = tid >> 1;        // 0..127 : tile row (A) / tile col (B)
    const int shalf = tid & 1;         // which 16-element half of BK=32

    // ---- precomputed staging addresses ----
    // LDS byte offsets (low 32 bits of the generic pointer = LDS offset)
    const unsigned ldsA = (unsigned)(uintptr_t)(void*)&As[0][srow * 40 + shalf * 16];
    const unsigned ldsB = (unsigned)(uintptr_t)(void*)&Bs[0][srow * 40 + shalf * 16];
    const int gcol  = tileN + srow;
    const int gcolC = (gcol < N) ? gcol : (N - 1);   // clamp: junk cols discarded at store
    const char* bGB = (const char*)(Bt + bOff + (long long)gcolC * ldb + shalf * 16);
    const float* aGf = (const float*)Av + aOff + (long long)(tileM + srow) * lda + shalf * 16;
    const char*  aGB = (const char*)((const f16*)Av + aOff + (long long)(tileM + srow) * lda + shalf * 16);

    const int nK = Kd >> 5;            // number of BK=32 slabs
    // async instructions issued per wave per slab
    constexpr int APER = (FLAGS & F_AF32) ? 2 : 4;

    auto stage = [&](int buf, int k0) {
        if (FLAGS & F_AF32) {
            // fp32 A (attention probs): load + convert + ds_store
            const v4f* s4 = (const v4f*)(aGf + k0);
            v4f f0 = s4[0], f1 = s4[1], f2 = s4[2], f3 = s4[3];
            v8h h0, h1;
            #pragma unroll
            for (int i = 0; i < 4; ++i) {
                h0[i]     = (f16)f0[i];  h0[4 + i] = (f16)f1[i];
                h1[i]     = (f16)f2[i];  h1[4 + i] = (f16)f3[i];
            }
            f16* dst = &As[buf][srow * 40 + shalf * 16];
            *(v8h*)dst = h0; *(v8h*)(dst + 8) = h1;
        } else {
            async_copy_32B(ldsA + buf * LDS_TILE_BYTES, aGB + 2 * (long long)k0);
        }
        async_copy_32B(ldsB + buf * LDS_TILE_BYTES, bGB + 2 * (long long)k0);
    };

    v8f acc[2][4] = {};

    // prologue: fill buffer 0
    stage(0, 0);

    for (int kt = 0; kt < nK; ++kt) {
        const int cur = kt & 1;
        if (kt + 1 < nK) {
            stage(cur ^ 1, (kt + 1) * 32);
            // drain the copies for the *current* buffer (in-order completion),
            // leaving the just-issued next-buffer copies in flight
            asm volatile("s_wait_asynccnt %0" :: "i"(APER) : "memory");
        } else {
            asm volatile("s_wait_asynccnt 0" ::: "memory");
        }
        __syncthreads();

        // ---- build fragments (ISA 7.12.2 16-bit layouts) ----
        const int r = lane & 15, g = lane >> 4;
        v16h af[2];
        #pragma unroll
        for (int m = 0; m < 2; ++m) {
            const int row = wM * 32 + m * 16 + r;
            union { v16h v; v8h h[2]; } u;
            // A lane(l): row = l%16; half g holds K = g*8..g*8+7 and 16+g*8..+7
            u.h[0] = *(const v8h*)&As[cur][row * 40 + g * 8];
            u.h[1] = *(const v8h*)&As[cur][row * 40 + 16 + g * 8];
            af[m] = u.v;
        }
        v16h bf[4];
        #pragma unroll
        for (int n = 0; n < 4; ++n) {
            const int col = wN * 64 + n * 16 + r;
            union { v16h v; v8h h[2]; } u;
            // B lane(l): col = l%16; half g holds K = g*8..g*8+7 and 16+g*8..+7
            u.h[0] = *(const v8h*)&Bs[cur][col * 40 + g * 8];
            u.h[1] = *(const v8h*)&Bs[cur][col * 40 + 16 + g * 8];
            bf[n] = u.v;
        }
        #pragma unroll
        for (int m = 0; m < 2; ++m)
            #pragma unroll
            for (int n = 0; n < 4; ++n)
                acc[m][n] = __builtin_amdgcn_wmma_f32_16x16x32_f16(
                    false, af[m], false, bf[n], (short)0, acc[m][n], false, false);
        __syncthreads();   // WAR: buffer `cur` is refilled next iteration
    }

    // ---- epilogue + store (C layout: VGPR r -> rows g*8+r, col = lane%16) ----
    const int rl = (lane >> 4) * 8;
    const int cl = lane & 15;
    #pragma unroll
    for (int m = 0; m < 2; ++m)
        #pragma unroll
        for (int n = 0; n < 4; ++n) {
            const int col = tileN + wN * 64 + n * 16 + cl;
            if (col >= N) continue;
            #pragma unroll
            for (int rr = 0; rr < 8; ++rr) {
                const int row = tileM + wM * 32 + m * 16 + rl + rr;
                float v = acc[m][n][rr];
                if (FLAGS & F_SCALE) v *= scale;
                if (FLAGS & F_BIAS)  v += bias[col];
                if (FLAGS & F_RESID) v += resid[rOff + (long long)row * ldr + col];
                if (FLAGS & F_GELU)  v = 0.5f * v * (1.0f + erff(v * 0.70710678118654752f));
                const long long ci = cOff + (long long)row * ldc + col;
                if (FLAGS & F_STF32) outF[ci] = v;
                if (FLAGS & F_STF16) outH[ci] = (f16)v;
            }
        }
}

// ---------------------------------------------------------------------------
// Row softmax with mask, in place on fp32 attn [NB*NH*TDIM rows x TDIM cols]
// ---------------------------------------------------------------------------
__global__ __launch_bounds__(256)
void softmax_mask(float* __restrict__ attn, const int* __restrict__ mask)
{
    __shared__ float red[256];
    const int rowId = blockIdx.x;                   // 0 .. NB*NH*TDIM-1
    const int b = rowId / (NH * TDIM);
    float* rowp = attn + (long long)rowId * TDIM;
    const int tid = threadIdx.x;

    float vals[8];
    float mx = -INFINITY;
    #pragma unroll
    for (int i = 0; i < 8; ++i) {
        const int c = tid + i * 256;
        float s = rowp[c];
        if (mask[b * TDIM + c] == 0) s = -INFINITY;
        vals[i] = s;
        mx = fmaxf(mx, s);
    }
    red[tid] = mx; __syncthreads();
    for (int s = 128; s > 0; s >>= 1) { if (tid < s) red[tid] = fmaxf(red[tid], red[tid + s]); __syncthreads(); }
    mx = red[0]; __syncthreads();

    float sum = 0.f;
    #pragma unroll
    for (int i = 0; i < 8; ++i) { float e = __expf(vals[i] - mx); vals[i] = e; sum += e; }
    red[tid] = sum; __syncthreads();
    for (int s = 128; s > 0; s >>= 1) { if (tid < s) red[tid] += red[tid + s]; __syncthreads(); }
    const float inv = 1.0f / red[0];
    #pragma unroll
    for (int i = 0; i < 8; ++i) rowp[tid + i * 256] = vals[i] * inv;
}

// ---------------------------------------------------------------------------
// Prep kernels
// ---------------------------------------------------------------------------
__global__ void cvt_f32_f16(const float* __restrict__ in, f16* __restrict__ out, long long n)
{
    long long i = (long long)blockIdx.x * blockDim.x + threadIdx.x;
    if (i < n) out[i] = (f16)in[i];
}

// wt[n*768 + k] = W{seg}[k*768 + n%768], n in [0,2304)
__global__ void pack_qkv(const float* __restrict__ Wq, const float* __restrict__ Wk,
                         const float* __restrict__ Wv, f16* __restrict__ wt)
{
    long long i = (long long)blockIdx.x * blockDim.x + threadIdx.x;
    if (i >= (long long)2304 * 768) return;
    const int n = (int)(i / 768), k = (int)(i % 768);
    const int seg = n / 768, col = n % 768;
    const float* W = (seg == 0) ? Wq : ((seg == 1) ? Wk : Wv);
    wt[i] = (f16)W[(long long)k * 768 + col];
}

__global__ void pack_bias3(const float* __restrict__ bq, const float* __restrict__ bk,
                           const float* __restrict__ bv, float* __restrict__ out)
{
    const int i = blockIdx.x * blockDim.x + threadIdx.x;
    if (i < 2304) out[i] = (i < 768) ? bq[i] : ((i < 1536) ? bk[i - 768] : bv[i - 1536]);
}

// wt[n*K + k] = (f16) W[k*N + n]   (W is [K][N] row-major)
__global__ void pack_t(const float* __restrict__ W, f16* __restrict__ wt, int K, int N)
{
    long long i = (long long)blockIdx.x * blockDim.x + threadIdx.x;
    if (i >= (long long)K * N) return;
    const int n = (int)(i / K), k = (int)(i % K);
    wt[i] = (f16)W[(long long)k * N + n];
}

// vt[(bh*64 + d)*2048 + t] = qkv[(b*2048 + t)*2304 + 1536 + h*64 + d]
__global__ void build_vt(const f16* __restrict__ qkv, f16* __restrict__ vt)
{
    long long i = (long long)blockIdx.x * blockDim.x + threadIdx.x;
    if (i >= (long long)NB * NH * HDIM * TDIM) return;
    const int t  = (int)(i % TDIM);
    long long j  = i / TDIM;
    const int d  = (int)(j % HDIM);
    const int bh = (int)(j / HDIM);
    const int b = bh / NH, h = bh % NH;
    vt[i] = qkv[((long long)(b * TDIM + t)) * 2304 + 1536 + h * HDIM + d];
}

// ---------------------------------------------------------------------------
extern "C" void kernel_launch(void* const* d_in, const int* in_sizes, int n_in,
                              void* d_out, int out_size, void* d_ws, size_t ws_size,
                              hipStream_t stream)
{
    const float* x    = (const float*)d_in[0];
    const int*   mask = (const int*)d_in[1];
    const float* Wq = (const float*)d_in[2];  const float* bq = (const float*)d_in[3];
    const float* Wk = (const float*)d_in[4];  const float* bk = (const float*)d_in[5];
    const float* Wv = (const float*)d_in[6];  const float* bv = (const float*)d_in[7];
    const float* Wo = (const float*)d_in[8];  const float* bo = (const float*)d_in[9];
    const float* W1 = (const float*)d_in[10]; const float* b1 = (const float*)d_in[11];
    const float* W2 = (const float*)d_in[12]; const float* b2 = (const float*)d_in[13];

    float* outp = (float*)d_out;                                 // [8192, 768]
    float* attn = outp + (long long)MROWS * BDIM;                // [48, 2048, 2048]

    // workspace carving (256-B aligned chunks)
    char* w = (char*)d_ws;
    auto alloc = [&](size_t bytes) -> void* {
        void* p = (void*)w;
        w += (bytes + 255) & ~(size_t)255;
        return p;
    };
    f16*   xh   = (f16*)  alloc((size_t)MROWS * BDIM * 2);
    f16*   wqkv = (f16*)  alloc((size_t)2304 * 768 * 2);
    float* bqkv = (float*)alloc((size_t)2304 * 4);
    f16*   wto  = (f16*)  alloc((size_t)768 * 768 * 2);
    f16*   wt1  = (f16*)  alloc((size_t)3072 * 768 * 2);
    f16*   wt2  = (f16*)  alloc((size_t)768 * 3072 * 2);
    f16*   qkv  = (f16*)  alloc((size_t)MROWS * 2304 * 2);
    f16*   vt   = (f16*)  alloc((size_t)NB * NH * HDIM * TDIM * 2);
    f16*   ctx  = (f16*)  alloc((size_t)MROWS * BDIM * 2);
    float* hf   = (float*)alloc((size_t)MROWS * BDIM * 4);
    f16*   hh   = (f16*)  alloc((size_t)MROWS * BDIM * 2);
    f16*   a1   = (f16*)  alloc((size_t)MROWS * DFF * 2);

    // --- prep ---
    {
        long long nx = (long long)MROWS * BDIM;
        cvt_f32_f16<<<(unsigned)((nx + 255) / 256), 256, 0, stream>>>(x, xh, nx);
        pack_qkv<<<(unsigned)(((long long)2304 * 768 + 255) / 256), 256, 0, stream>>>(Wq, Wk, Wv, wqkv);
        pack_bias3<<<9, 256, 0, stream>>>(bq, bk, bv, bqkv);
        pack_t<<<(unsigned)(((long long)768 * 768 + 255) / 256), 256, 0, stream>>>(Wo, wto, 768, 768);
        pack_t<<<(unsigned)(((long long)768 * 3072 + 255) / 256), 256, 0, stream>>>(W1, wt1, 768, 3072);
        pack_t<<<(unsigned)(((long long)3072 * 768 + 255) / 256), 256, 0, stream>>>(W2, wt2, 3072, 768);
    }

    // --- GEMM1: QKV = xh @ wqkv^T + bqkv -> f16 [8192, 2304] ---
    {
        dim3 g(2304 / 128, MROWS / 128, 1);
        gemm_wmma<F_BIAS | F_STF16><<<g, 256, 0, stream>>>(
            xh, 0, 0, BDIM, wqkv, 0, 0, BDIM, bqkv,
            nullptr, 0, 0, 0, nullptr, qkv, 0, 0, 2304, 2304, BDIM, 1, 1.0f);
    }

    // --- V transpose: vt[bh][d][t] ---
    build_vt<<<(unsigned)(((long long)NB * NH * HDIM * TDIM + 255) / 256), 256, 0, stream>>>(qkv, vt);

    // --- GEMM2: scores = (Q @ K^T) * 1/sqrt(64) -> fp32 attn region of d_out ---
    {
        dim3 g(TDIM / 128, TDIM / 128, NB * NH);
        gemm_wmma<F_SCALE | F_STF32><<<g, 256, 0, stream>>>(
            qkv,        (long long)TDIM * 2304, 64, 2304,          // Q section
            qkv + 768,  (long long)TDIM * 2304, 64, 2304,          // K section (Bt = K, N-major)
            nullptr, nullptr, 0, 0, 0,
            attn, nullptr, (long long)NH * TDIM * TDIM, (long long)TDIM * TDIM, TDIM,
            TDIM, HDIM, NH, 0.125f);
    }

    // --- masked softmax in place on attn ---
    softmax_mask<<<NB * NH * TDIM, 256, 0, stream>>>(attn, mask);

    // --- GEMM3: ctx = attn(fp32->f16) @ V -> f16 [8192, 768] (per-head cols) ---
    {
        dim3 g(1, TDIM / 128, NB * NH);
        gemm_wmma<F_AF32 | F_STF16><<<g, 256, 0, stream>>>(
            attn, (long long)NH * TDIM * TDIM, (long long)TDIM * TDIM, TDIM,
            vt,   (long long)NH * HDIM * TDIM, (long long)HDIM * TDIM, TDIM,
            nullptr, nullptr, 0, 0, 0,
            nullptr, ctx, (long long)TDIM * BDIM, HDIM, BDIM,
            HDIM, TDIM, NH, 1.0f);
    }

    // --- GEMM4: h = x + ctx @ Wo^T + bo -> hf (f32) and hh (f16) ---
    {
        dim3 g(BDIM / 128, MROWS / 128, 1);
        gemm_wmma<F_BIAS | F_RESID | F_STF32 | F_STF16><<<g, 256, 0, stream>>>(
            ctx, 0, 0, BDIM, wto, 0, 0, BDIM, bo,
            x, 0, 0, BDIM, hf, hh, 0, 0, BDIM, BDIM, BDIM, 1, 1.0f);
    }

    // --- GEMM5: a1 = gelu(h @ W1^T + b1) -> f16 [8192, 3072] ---
    {
        dim3 g(DFF / 128, MROWS / 128, 1);
        gemm_wmma<F_BIAS | F_GELU | F_STF16><<<g, 256, 0, stream>>>(
            hh, 0, 0, BDIM, wt1, 0, 0, BDIM, b1,
            nullptr, 0, 0, 0, nullptr, a1, 0, 0, DFF, DFF, BDIM, 1, 1.0f);
    }

    // --- GEMM6: out = h + a1 @ W2^T + b2 -> fp32 out region of d_out ---
    {
        dim3 g(BDIM / 128, MROWS / 128, 1);
        gemm_wmma<F_BIAS | F_RESID | F_STF32><<<g, 256, 0, stream>>>(
            a1, 0, 0, DFF, wt2, 0, 0, DFF, b2,
            hf, 0, 0, BDIM, outp, nullptr, 0, 0, BDIM, BDIM, DFF, 1, 1.0f);
    }
}
